// DyConv2d_58574763983353
// MI455X (gfx1250) — compile-verified, matrix operand order checked by reading, not run on hardware
//
#include <hip/hip_runtime.h>
#include <hip/hip_bf16.h>

// ---- CDNA5 WMMA vector types ----
typedef __attribute__((ext_vector_type(16))) __bf16        v16bf;
typedef __attribute__((ext_vector_type(2)))  __bf16        v2bf;
typedef __attribute__((ext_vector_type(8)))  float         v8f;
typedef __attribute__((ext_vector_type(8)))  unsigned int  v8u;

#define C_IN  64
#define HW    65536          // 256*256
#define NMID  17
#define NEXP  4
#define TEMP  34.0f

// fp32 -> bf16 bits via native bfloat type (RNE fptrunc)
__device__ __forceinline__ unsigned short f2bf(float f) {
  __bf16 h = (__bf16)f;
  return __builtin_bit_cast(unsigned short, h);
}

// two fp32 -> packed bf16x2 (low = f0, high = f1)
__device__ __forceinline__ unsigned int pack_bf16(float f0, float f1) {
#if __has_builtin(__builtin_amdgcn_cvt_pk_bf16_f32)
  v2bf p = __builtin_amdgcn_cvt_pk_bf16_f32(f0, f1);
  return __builtin_bit_cast(unsigned int, p);
#else
  v2bf p = { (__bf16)f0, (__bf16)f1 };   // let ISel pick v_cvt_pk_bf16_f32
  return __builtin_bit_cast(unsigned int, p);
#endif
}

// ---------------------------------------------------------------------------
// Kernel 1: global average pool.  One block per (b,c) plane of 65536 floats.
// ---------------------------------------------------------------------------
__global__ void __launch_bounds__(256) pool_kernel(const float* __restrict__ x,
                                                   float* __restrict__ pooled) {
  const int plane = blockIdx.x;                       // b*64 + c
  const float4* p = (const float4*)(x + (size_t)plane * HW);
  float s = 0.f;
  for (int i = threadIdx.x; i < HW / 4; i += 256) {
    float4 v = p[i];
    s += v.x + v.y + v.z + v.w;
  }
  __shared__ float red[256];
  red[threadIdx.x] = s;
  __syncthreads();
  for (int off = 128; off > 0; off >>= 1) {
    if (threadIdx.x < off) red[threadIdx.x] += red[threadIdx.x + off];
    __syncthreads();
  }
  if (threadIdx.x == 0) pooled[plane] = red[0] * (1.0f / 65536.0f);
}

// ---------------------------------------------------------------------------
// Kernel 2: channel attention MLP + softmax + expert mixing.
// One block (64 threads) per batch sample. Produces bf16 W[b][64][64]
// (row-major: o major, i minor) and fp32 bias[b][64].
// ---------------------------------------------------------------------------
__global__ void __launch_bounds__(64) attn_mix_kernel(
    const float* __restrict__ pooled,
    const float* __restrict__ fc1_w,     // [17,64]
    const float* __restrict__ fc2_w,     // [4,17]
    const float* __restrict__ fc2_b,     // [4]
    const float* __restrict__ expert_w,  // [4,64,64]
    const float* __restrict__ expert_b,  // [4,64]
    unsigned short* __restrict__ wmix,   // [8,64,64] bf16 bits
    float* __restrict__ bias) {          // [8,64]
  const int b = blockIdx.x;
  const int t = threadIdx.x;
  __shared__ float hmid[NMID];
  __shared__ float logits[NEXP];
  __shared__ float attn[NEXP];

  if (t < NMID) {
    float s = 0.f;
    for (int c = 0; c < C_IN; ++c) s += pooled[b * C_IN + c] * fc1_w[t * C_IN + c];
    hmid[t] = fmaxf(s, 0.f);
  }
  __syncthreads();
  if (t < NEXP) {
    float s = fc2_b[t];
    for (int m = 0; m < NMID; ++m) s += hmid[m] * fc2_w[t * NMID + m];
    logits[t] = s * (1.0f / TEMP);
  }
  __syncthreads();
  if (t == 0) {
    float mx = logits[0];
    for (int e = 1; e < NEXP; ++e) mx = fmaxf(mx, logits[e]);
    float ex[NEXP], den = 0.f;
    for (int e = 0; e < NEXP; ++e) { ex[e] = __expf(logits[e] - mx); den += ex[e]; }
    for (int e = 0; e < NEXP; ++e) attn[e] = ex[e] / den;
  }
  __syncthreads();

  const int o = t;
  float a[NEXP];
  for (int e = 0; e < NEXP; ++e) a[e] = attn[e];

  float bs = 0.f;
  for (int e = 0; e < NEXP; ++e) bs += a[e] * expert_b[e * C_IN + o];
  bias[b * C_IN + o] = bs;

  for (int i = 0; i < C_IN; ++i) {
    float wv = 0.f;
    for (int e = 0; e < NEXP; ++e) wv += a[e] * expert_w[(e * C_IN + o) * C_IN + i];
    wmix[(b * C_IN + o) * C_IN + i] = f2bf(wv);
  }
}

// ---------------------------------------------------------------------------
// Kernel 3: per-sample 1x1 conv as GEMM with bf16 WMMA.
//   out[b] (64 x 65536) = W[b] (64x64) * X[b] (64x65536) + bias[b]
// Block = 256 threads = 8 waves; each wave computes a 64(M) x 16(N) strip.
// Bias pre-loaded into WMMA C accumulator. half/k-step folded into base
// pointers so all channel strides are 24-bit immediate offsets (<= 5.75 MB).
// Grid = (65536/128, 8).
// ---------------------------------------------------------------------------
__global__ void __launch_bounds__(256) dyconv_kernel(
    const float* __restrict__ x,
    const unsigned short* __restrict__ wmix,
    const float* __restrict__ bias,
    float* __restrict__ out) {
  const int lane = threadIdx.x & 31;
  const int wave = threadIdx.x >> 5;
  const int half = lane >> 4;           // 0: lanes 0-15, 1: lanes 16-31
  const int n    = lane & 15;           // column / row-within-tile index
  const int b    = blockIdx.y;
  const size_t pix = (size_t)blockIdx.x * 128 + (size_t)wave * 16 + n;

  const float* xb = x + (size_t)b * C_IN * HW;
  float*       ob = out + (size_t)b * C_IN * HW;
  const unsigned short* Wb    = wmix + b * (C_IN * C_IN);
  const float*          biasb = bias + b * C_IN;

  // --- A fragments (mixed-weight matrix, bf16), register resident ---
  // 16-bit A 16x32 layout: lanes 0-15 row M=lane: VGPR0-3 K=0..7, VGPR4-7 K=16..23
  //                        lanes 16-31 row M=lane-16: K=8..15 and K=24..31
  // Base folds lane row (n) and half; remaining offsets are small immediates.
  const unsigned short* Wbase = Wb + n * C_IN + half * 8;
  v16bf afrag[2][4];
#pragma unroll
  for (int ks = 0; ks < 2; ++ks) {
#pragma unroll
    for (int mt = 0; mt < 4; ++mt) {
      const uint4 a0 = *(const uint4*)(Wbase + mt * 16 * C_IN + ks * 32);      // K lo..lo+7
      const uint4 a1 = *(const uint4*)(Wbase + mt * 16 * C_IN + ks * 32 + 16); // K lo+16..lo+23
      v8u tmp = { a0.x, a0.y, a0.z, a0.w, a1.x, a1.y, a1.z, a1.w };
      afrag[ks][mt] = __builtin_bit_cast(v16bf, tmp);
    }
  }

  // --- C accumulators pre-loaded with bias ---
  // C/D layout: VGPR r -> M = mt*16 + half*8 + r (uniform across lanes)
  const float* bbias = biasb + half * 8;
  v8f acc[4];
#pragma unroll
  for (int mt = 0; mt < 4; ++mt) {
#pragma unroll
    for (int r = 0; r < 8; ++r) {
      acc[mt][r] = bbias[mt * 16 + r];
    }
  }

  // prefetch next block's slice of x (stream-ahead into GL2)
  __builtin_prefetch(xb + pix + 128, 0, 1);

  // Base pointer per k-step: folds pix, half*8, ks*32 -> channel offsets
  // in the loads are constant immediates (max 23 * 256KB = 5.75 MB).
  const float* bbase0 = xb + pix + (size_t)(half * 8) * HW;

#pragma unroll
  for (int ks = 0; ks < 2; ++ks) {
    const float* bb = bbase0 + (size_t)(ks * 32) * HW;
    // --- build B fragment (X columns), mirrored A layout: lane = column N ---
    unsigned int pk[8];
#pragma unroll
    for (int j = 0; j < 8; ++j) {
      const int koff = (j < 4) ? (2 * j) : (16 + 2 * (j - 4));
      const float f0 = bb[(size_t)koff * HW];
      const float f1 = bb[(size_t)(koff + 1) * HW];
      pk[j] = pack_bf16(f0, f1);
    }
    v8u bu = { pk[0], pk[1], pk[2], pk[3], pk[4], pk[5], pk[6], pk[7] };
    v16bf bfrag = __builtin_bit_cast(v16bf, bu);

    // 4 M-tiles share the B fragment
    acc[0] = __builtin_amdgcn_wmma_f32_16x16x32_bf16(
        false, afrag[ks][0], false, bfrag, (short)0, acc[0], false, false);
    acc[1] = __builtin_amdgcn_wmma_f32_16x16x32_bf16(
        false, afrag[ks][1], false, bfrag, (short)0, acc[1], false, false);
    acc[2] = __builtin_amdgcn_wmma_f32_16x16x32_bf16(
        false, afrag[ks][2], false, bfrag, (short)0, acc[2], false, false);
    acc[3] = __builtin_amdgcn_wmma_f32_16x16x32_bf16(
        false, afrag[ks][3], false, bfrag, (short)0, acc[3], false, false);
  }

  // --- epilogue: store D directly (bias already inside accumulator) ---
  // Non-temporal stores keep x resident in L2; one pointer bump per M-tile,
  // row offsets are constant immediates (max 7 * 256KB).
  float* ob0 = ob + pix + (size_t)(half * 8) * HW;
#pragma unroll
  for (int mt = 0; mt < 4; ++mt) {
    float* obm = ob0 + (size_t)(mt * 16) * HW;
#pragma unroll
    for (int r = 0; r < 8; ++r) {
      __builtin_nontemporal_store(acc[mt][r], obm + (size_t)r * HW);
    }
  }
}

// ---------------------------------------------------------------------------
extern "C" void kernel_launch(void* const* d_in, const int* in_sizes, int n_in,
                              void* d_out, int out_size, void* d_ws, size_t ws_size,
                              hipStream_t stream) {
  const float* x        = (const float*)d_in[0];  // [8,64,256,256]
  const float* fc1_w    = (const float*)d_in[1];  // [17,64]
  const float* fc2_w    = (const float*)d_in[2];  // [4,17]
  const float* fc2_b    = (const float*)d_in[3];  // [4]
  const float* expert_w = (const float*)d_in[4];  // [4,64,64,1,1]
  const float* expert_b = (const float*)d_in[5];  // [4,64]
  float* out = (float*)d_out;

  // workspace layout
  float*          pooled  = (float*)d_ws;                     // 512 floats
  float*          biasbuf = pooled + 512;                     // 512 floats
  unsigned short* wmixbuf = (unsigned short*)(biasbuf + 512); // 8*4096 bf16

  pool_kernel<<<8 * C_IN, 256, 0, stream>>>(x, pooled);
  attn_mix_kernel<<<8, 64, 0, stream>>>(pooled, fc1_w, fc2_w, fc2_b,
                                        expert_w, expert_b, wmixbuf, biasbuf);
  dim3 grid(HW / 128, 8);
  dyconv_kernel<<<grid, 256, 0, stream>>>(x, wmixbuf, biasbuf, out);
}
